// SSMNode_7043746365633
// MI455X (gfx1250) — matrix operand order repaired and checked by reading, not compiled, not run on hardware
//
#include <hip/hip_runtime.h>
#include <hip/hip_bf16.h>
#include <math.h>

typedef __bf16 bf16_t;
typedef __attribute__((ext_vector_type(16))) __bf16 v16bf;
typedef __attribute__((ext_vector_type(8)))  __bf16 v8bf;
typedef __attribute__((ext_vector_type(8)))  float  v8f;

#define LDS_STRIDE 40   // 32 K elements + 8 pad (row = 80B, 16B-aligned segments)

// ---------------------------------------------------------------------------
// CDNA5 async global->LDS copy (ASYNCcnt-tracked), GVS addressing:
//   mem = SGPR64(base) + VGPR32(gl_off) ; LDS dest = wave LDS base + lds_off
// ---------------------------------------------------------------------------
__device__ __forceinline__ void async_ld_b128(unsigned lds_off, unsigned gl_off,
                                              const void* base)
{
    asm volatile("global_load_async_to_lds_b128 %0, %1, %2"
                 :: "v"(lds_off), "v"(gl_off), "s"(base)
                 : "memory");
}
__device__ __forceinline__ void wait_async0()
{
    asm volatile("s_wait_asynccnt 0x0" ::: "memory");
}

// ---------------------------------------------------------------------------
// RMSNorm: one block per row, writes bf16 normalized activations
// ---------------------------------------------------------------------------
__global__ __launch_bounds__(256) void rmsnorm_bf16(
    const float* __restrict__ x, const float* __restrict__ w,
    bf16_t* __restrict__ out, int D)
{
    int row = blockIdx.x;
    const float* xr = x + (size_t)row * D;
    float ss = 0.f;
    for (int i = threadIdx.x; i < D; i += 256) { float v = xr[i]; ss += v * v; }
    __shared__ float red[256];
    red[threadIdx.x] = ss;
    __syncthreads();
    for (int s = 128; s > 0; s >>= 1) {
        if (threadIdx.x < s) red[threadIdx.x] += red[threadIdx.x + s];
        __syncthreads();
    }
    float scale = rsqrtf(red[0] / (float)D + 1e-6f);
    for (int i = threadIdx.x; i < D; i += 256)
        out[(size_t)row * D + i] = (bf16_t)(xr[i] * scale * w[i]);
}

// ---------------------------------------------------------------------------
// fp32 -> bf16 conversion
// ---------------------------------------------------------------------------
__global__ void cvt_bf16(const float* __restrict__ in, bf16_t* __restrict__ out, int n)
{
    int i = blockIdx.x * blockDim.x + threadIdx.x;
    if (i < n) out[i] = (bf16_t)in[i];
}

// ---------------------------------------------------------------------------
// WMMA bf16 GEMM: C[M,N] = A[M,K] @ B[K,N], fp32 accumulate.
// Compile-time N, K (all strides fold into instruction offsets).
// Block tile 128x128, BK=32, 8 waves (4x2), each wave 32x64 = 2x4 WMMA tiles.
// Double-buffered LDS; A tile staged with global_load_async_to_lds_b128,
// B tile transposed through VGPRs into [n][k] layout.
// EPI==0: plain fp32 store. EPI==1: softplus(x + bias[n]).
// Requires M%128==0, N%128==0, K%32==0 (true for all calls here).
// ---------------------------------------------------------------------------
template <int EPI, int N, int K>
__global__ __launch_bounds__(256) void gemm_bf16_wmma(
    const bf16_t* __restrict__ A, const bf16_t* __restrict__ Bw,
    float* __restrict__ C, const float* __restrict__ bias, int M)
{
    __shared__ bf16_t ldsA[2][128 * LDS_STRIDE];   // [m][k]
    __shared__ bf16_t ldsB[2][128 * LDS_STRIDE];   // [n][k] (transposed)

    const int tid  = threadIdx.x;
    const int lane = tid & 31;
    const int wave = tid >> 5;
    const int wm   = wave >> 1;        // 0..3  -> 32 rows each
    const int wn   = wave & 1;         // 0..1  -> 64 cols each
    const int blockM = blockIdx.y * 128;
    const int blockN = blockIdx.x * 128;

    v8f acc[2][4];
#pragma unroll
    for (int i = 0; i < 2; ++i)
#pragma unroll
        for (int j = 0; j < 4; ++j)
            acc[i][j] = (v8f){0.f,0.f,0.f,0.f,0.f,0.f,0.f,0.f};

    const int lrow = lane & 15;
    const int kbA  = (lane < 16) ? 0 : 8;    // A 16x32 bf16 lane layout
    const int kbB  = (lane < 16) ? 0 : 8;    // B 32x16 bf16 lane layout

    // ---- tile staging (A: async to LDS, B: transposed through VGPRs) ----
    auto stage = [&](int buf, int k0) {
#pragma unroll
        for (int c = 0; c < 2; ++c) {              // A: 512 16B chunks / block
            int chunk = tid * 2 + c;
            int row = chunk >> 2;
            int seg = chunk & 3;
            unsigned lds_off = (unsigned)(uintptr_t)&ldsA[buf][row * LDS_STRIDE + seg * 8];
            unsigned gl_off  = (unsigned)(((blockM + row) * K + k0 + seg * 8) * 2);
            async_ld_b128(lds_off, gl_off, A);
        }
#pragma unroll
        for (int c = 0; c < 2; ++c) {              // B: 32x128 -> [n][k]
            int chunk = tid * 2 + c;
            int krow = chunk >> 4;
            int nseg = chunk & 15;
            v8bf v = *(const v8bf*)(Bw + (size_t)(k0 + krow) * N + blockN + nseg * 8);
#pragma unroll
            for (int e = 0; e < 8; ++e)
                ldsB[buf][(nseg * 8 + e) * LDS_STRIDE + krow] = v[e];
        }
    };

    stage(0, 0);

    const int NTILE = K / 32;
    for (int it = 0; it < NTILE; ++it) {
        const int buf = it & 1;
        wait_async0();          // my async A-chunks for this buffer are in LDS
        __syncthreads();        // (also waits my B ds_stores); all staging done

        if (it + 1 < NTILE) stage(buf ^ 1, (it + 1) * 32);

        // ---- load fragments from LDS ----
        v16bf afrag[2], bfrag[4];
#pragma unroll
        for (int i = 0; i < 2; ++i) {
            const bf16_t* base = &ldsA[buf][(wm * 32 + i * 16 + lrow) * LDS_STRIDE + kbA];
            v8bf lo = *(const v8bf*)(base);
            v8bf hi = *(const v8bf*)(base + 16);
            v16bf a;
#pragma unroll
            for (int e = 0; e < 8; ++e) { a[e] = lo[e]; a[e + 8] = hi[e]; }
            afrag[i] = a;
        }
#pragma unroll
        for (int j = 0; j < 4; ++j) {
            const bf16_t* base = &ldsB[buf][(wn * 64 + j * 16 + lrow) * LDS_STRIDE + kbB];
            v8bf lo = *(const v8bf*)(base);
            v8bf hi = *(const v8bf*)(base + 16);
            v16bf b;
#pragma unroll
            for (int e = 0; e < 8; ++e) { b[e] = lo[e]; b[e + 8] = hi[e]; }
            bfrag[j] = b;
        }

        // ---- 8 WMMAs ----
#pragma unroll
        for (int i = 0; i < 2; ++i)
#pragma unroll
            for (int j = 0; j < 4; ++j)
                acc[i][j] = __builtin_amdgcn_wmma_f32_16x16x32_bf16(
                    false, afrag[i], false, bfrag[j],
                    (short)0, acc[i][j], false, false);
    }

    // ---- epilogue: D VGPR r -> row = r + (lane<16 ? 0 : 8), col = lane&15 ----
    const int rbase = (lane < 16) ? 0 : 8;
#pragma unroll
    for (int i = 0; i < 2; ++i) {
#pragma unroll
        for (int j = 0; j < 4; ++j) {
            int col  = blockN + wn * 64 + j * 16 + lrow;
            int row0 = blockM + wm * 32 + i * 16 + rbase;
            float* cp = C + (size_t)row0 * N + col;
#pragma unroll
            for (int r = 0; r < 8; ++r) {
                float v = acc[i][j][r];
                if (EPI == 1) {
                    v += bias[col];
                    v = (v > 20.f) ? v : log1pf(__expf(v));
                }
                cp[r * N] = v;     // constant N -> immediate store offsets
            }
        }
    }
}

// ---------------------------------------------------------------------------
// x_inner = a * silu(g),  a = xg[:, :di], g = xg[:, di:]
// ---------------------------------------------------------------------------
__global__ void gated_silu(const float* __restrict__ xg, float* __restrict__ xi,
                           int rows, int di)
{
    int idx = blockIdx.x * blockDim.x + threadIdx.x;
    if (idx >= rows * di) return;
    int m = idx / di, j = idx - m * di;
    float a = xg[(size_t)m * 2 * di + j];
    float g = xg[(size_t)m * 2 * di + di + j];
    xi[idx] = a * (g / (1.f + __expf(-g)));
}

// ---------------------------------------------------------------------------
// causal depthwise conv, kernel 4 (left pad 3): xc fp32 + bf16
// ---------------------------------------------------------------------------
__global__ void dwconv_causal(const float* __restrict__ xi, const float* __restrict__ w,
                              float* __restrict__ xc, bf16_t* __restrict__ xc_bf,
                              int Bsz, int T, int DI)
{
    int idx = blockIdx.x * blockDim.x + threadIdx.x;
    if (idx >= Bsz * T * DI) return;
    int c = idx % DI;
    int r = idx / DI;       // r = b*T + t
    int t = r % T;
    float acc = 0.f;
#pragma unroll
    for (int j = 0; j < 4; ++j) {
        int tt = t - 3 + j;
        if (tt >= 0) acc += w[c * 4 + j] * xi[(size_t)(r - 3 + j) * DI + c];
    }
    xc[idx] = acc;
    xc_bf[idx] = (bf16_t)acc;
}

// ---------------------------------------------------------------------------
// Bm = xc @ B_w,  Cm = xc @ C_w  (N = 16 each): block per row, LDS atomics
// ---------------------------------------------------------------------------
__global__ __launch_bounds__(128) void bc_proj(
    const float* __restrict__ xc, const float* __restrict__ Bw,
    const float* __restrict__ Cw, float* __restrict__ Bm, float* __restrict__ Cm,
    int Kd)
{
    __shared__ float sacc[32];
    int m = blockIdx.x, tid = threadIdx.x;
    if (tid < 32) sacc[tid] = 0.f;
    __syncthreads();
    float accB[16] = {0.f}, accC[16] = {0.f};
    for (int k = tid; k < Kd; k += 128) {
        float xv = xc[(size_t)m * Kd + k];
#pragma unroll
        for (int s = 0; s < 16; ++s) {
            accB[s] += xv * Bw[k * 16 + s];
            accC[s] += xv * Cw[k * 16 + s];
        }
    }
#pragma unroll
    for (int s = 0; s < 16; ++s) {
        atomicAdd(&sacc[s], accB[s]);
        atomicAdd(&sacc[16 + s], accC[s]);
    }
    __syncthreads();
    if (tid < 16) {
        Bm[(size_t)m * 16 + tid] = sacc[tid];
        Cm[(size_t)m * 16 + tid] = sacc[16 + tid];
    }
}

// ---------------------------------------------------------------------------
// Sequential selective scan over T; one lane per (b, d), 16 states in regs.
// ---------------------------------------------------------------------------
__global__ __launch_bounds__(256) void ssm_scan(
    const float* __restrict__ dt, const float* __restrict__ xc,
    const float* __restrict__ Bm, const float* __restrict__ Cm,
    const float* __restrict__ A_log, const float* __restrict__ D_skip,
    bf16_t* __restrict__ y_bf, int T, int DI)
{
    int blocksPerB = DI / 256;
    int b = blockIdx.x / blocksPerB;
    int d = (blockIdx.x % blocksPerB) * 256 + threadIdx.x;

    float Arow[16], h[16];
#pragma unroll
    for (int s = 0; s < 16; ++s) {
        Arow[s] = -__expf(A_log[(size_t)d * 16 + s]);
        h[s] = 0.f;
    }
    float Dsk = D_skip[d];

    for (int t = 0; t < T; ++t) {
        size_t row = (size_t)b * T + t;
        float dtv = dt[row * DI + d];
        float xcv = xc[row * DI + d];
        const float* Bp = Bm + row * 16;
        const float* Cp = Cm + row * 16;
        float y = Dsk * xcv;
#pragma unroll
        for (int s = 0; s < 16; ++s) {
            float a = __expf(dtv * Arow[s]);
            h[s] = a * h[s] + dtv * Bp[s] * xcv;
            y += h[s] * Cp[s];
        }
        y_bf[row * DI + d] = (bf16_t)y;
    }
}

// ---------------------------------------------------------------------------
// Host launcher
// ---------------------------------------------------------------------------
extern "C" void kernel_launch(void* const* d_in, const int* in_sizes, int n_in,
                              void* d_out, int out_size, void* d_ws, size_t ws_size,
                              hipStream_t stream)
{
    const int Bsz = 2, T = 1024, DM = 1024, DI = 2048, DS = 16;
    const int M = Bsz * T;                 // 2048 rows

    const float* x         = (const float*)d_in[0];
    const float* norm_w    = (const float*)d_in[1];
    const float* in_proj_w = (const float*)d_in[2];   // [DM, 2*DI]
    const float* conv_w    = (const float*)d_in[3];   // [DI, 1, 4]
    const float* dt_w      = (const float*)d_in[4];   // [DI, DI]
    const float* dt_b      = (const float*)d_in[5];   // [DI]
    const float* A_log     = (const float*)d_in[6];   // [DI, DS]
    const float* B_w       = (const float*)d_in[7];   // [DI, DS]
    const float* C_w       = (const float*)d_in[8];   // [DI, DS]
    const float* D_skip    = (const float*)d_in[9];   // [DI]
    const float* out_proj_w= (const float*)d_in[10];  // [DI, DM]
    float* out = (float*)d_out;

    // bump allocator over workspace
    size_t off = 0;
    auto alloc = [&](size_t bytes) -> void* {
        void* p = (char*)d_ws + off;
        off += (bytes + 255) & ~(size_t)255;
        return p;
    };
    bf16_t* xn_bf     = (bf16_t*)alloc((size_t)M * DM * 2);
    bf16_t* w_in_bf   = (bf16_t*)alloc((size_t)DM * 2 * DI * 2);
    bf16_t* w_dt_bf   = (bf16_t*)alloc((size_t)DI * DI * 2);
    bf16_t* w_out_bf  = (bf16_t*)alloc((size_t)DI * DM * 2);
    float*  xg        = (float*) alloc((size_t)M * 2 * DI * 4);
    float*  xinner    = (float*) alloc((size_t)M * DI * 4);
    float*  xc        = (float*) alloc((size_t)M * DI * 4);
    bf16_t* xc_bf     = (bf16_t*)alloc((size_t)M * DI * 2);
    float*  dt        = (float*) alloc((size_t)M * DI * 4);
    float*  Bm        = (float*) alloc((size_t)M * DS * 4);
    float*  Cm        = (float*) alloc((size_t)M * DS * 4);
    bf16_t* y_bf      = (bf16_t*)alloc((size_t)M * DI * 2);
    (void)ws_size;

    // 1) RMSNorm -> bf16
    rmsnorm_bf16<<<M, 256, 0, stream>>>(x, norm_w, xn_bf, DM);

    // 2) weight conversions fp32 -> bf16
    {
        int n1 = DM * 2 * DI, n2 = DI * DI, n3 = DI * DM;
        cvt_bf16<<<(n1 + 255) / 256, 256, 0, stream>>>(in_proj_w, w_in_bf, n1);
        cvt_bf16<<<(n2 + 255) / 256, 256, 0, stream>>>(dt_w,      w_dt_bf, n2);
        cvt_bf16<<<(n3 + 255) / 256, 256, 0, stream>>>(out_proj_w,w_out_bf,n3);
    }

    // 3) in_proj GEMM: [2048,1024] @ [1024,4096]
    gemm_bf16_wmma<0, 4096, 1024><<<dim3(4096 / 128, M / 128), 256, 0, stream>>>(
        xn_bf, w_in_bf, xg, nullptr, M);

    // 4) gated SiLU
    {
        int n = M * DI;
        gated_silu<<<(n + 255) / 256, 256, 0, stream>>>(xg, xinner, M, DI);
    }

    // 5) causal depthwise conv
    {
        int n = Bsz * T * DI;
        dwconv_causal<<<(n + 255) / 256, 256, 0, stream>>>(xinner, conv_w, xc, xc_bf, Bsz, T, DI);
    }

    // 6) dt GEMM with fused softplus+bias: [2048,2048] @ [2048,2048]
    gemm_bf16_wmma<1, 2048, 2048><<<dim3(2048 / 128, M / 128), 256, 0, stream>>>(
        xc_bf, w_dt_bf, dt, dt_b, M);

    // 7) B/C projections (N=16)
    bc_proj<<<M, 128, 0, stream>>>(xc, B_w, C_w, Bm, Cm, DI);

    // 8) sequential scan over T
    ssm_scan<<<Bsz * (DI / 256), 256, 0, stream>>>(dt, xc, Bm, Cm, A_log, D_skip, y_bf, T, DI);

    // 9) out_proj GEMM -> final output: [2048,2048] @ [2048,1024]
    gemm_bf16_wmma<0, 1024, 2048><<<dim3(1024 / 128, M / 128), 256, 0, stream>>>(
        y_bf, w_out_bf, out, nullptr, M);
}
